// QLoRAAdapter_74998718923208
// MI455X (gfx1250) — matrix operand order repaired and checked by reading, not compiled, not run on hardware
//
#include <hip/hip_runtime.h>

typedef __attribute__((ext_vector_type(16))) _Float16 v16h;
typedef __attribute__((ext_vector_type(8)))  float    v8f;

#define M_TOT (8 * 8192)
#define N_TOT 1024
#define K_TOT 1024
#define BM 128
#define BN 128
#define BK 32
#define ASTR 40            // padded LDS row stride in halfs (80B, 16B-aligned)
#define NTHREADS 256

// ---------------------------------------------------------------------------
// Pre-pass: W_eff[n][k] = qW[n][k]*w_scale[n] + 2.0 * sum_r B[n][r]*A[r][k]
// stored as f16, row-major [N][K] so GEMM B-fragments are K-contiguous.
// ---------------------------------------------------------------------------
__global__ __launch_bounds__(256) void weff_kernel(
    const int* __restrict__ qW, const float* __restrict__ wscale,
    const float* __restrict__ Amat, const float* __restrict__ Bmat,
    _Float16* __restrict__ Weff)
{
    const int k = blockIdx.x * blockDim.x + threadIdx.x;  // 0..1023
    const int n = blockIdx.y;                             // 0..1023
    float w = (float)qW[n * K_TOT + k] * wscale[n];
    float lo = 0.f;
#pragma unroll
    for (int r = 0; r < 16; ++r)
        lo += Bmat[n * 16 + r] * Amat[r * K_TOT + k];
    Weff[(size_t)n * K_TOT + k] = (_Float16)(w + 2.0f * lo);
}

// ---------------------------------------------------------------------------
// Main GEMM: out[m][n] = sum_k x[m][k] * Weff[n][k] + bias[n]
// 256 threads = 8 waves (wave32). Block tile 128x128, K-step 32.
// Wave grid 2(M) x 4(N); each wave owns 4x2 WMMA 16x16 tiles.
// B (Weff) tile is copied global->LDS with GLOBAL_LOAD_ASYNC_TO_LDS_B128
// (pure f16 copy, no VGPR staging, tracked by ASYNCcnt).
// ---------------------------------------------------------------------------
__global__ __launch_bounds__(NTHREADS) void qlora_gemm(
    const float* __restrict__ x, const _Float16* __restrict__ Weff,
    const float* __restrict__ bias, float* __restrict__ out)
{
    __shared__ __align__(16) _Float16 As[2][BM * ASTR];
    __shared__ __align__(16) _Float16 Bs[2][BN * ASTR];

    const int tid   = threadIdx.x;
    const int lane  = tid & 31;
    const int wave  = tid >> 5;
    const int waveM = wave & 1;    // 0..1  -> 64-row strip
    const int waveN = wave >> 1;   // 0..3  -> 32-col strip
    const int col   = lane & 15;   // matrix row (A) / col (B,C)
    const int hi    = lane >> 4;   // half-wave select

    const int m0 = blockIdx.y * BM;
    const int n0 = blockIdx.x * BN;

    v8f acc[4][2];
#pragma unroll
    for (int i = 0; i < 4; ++i)
#pragma unroll
        for (int j = 0; j < 2; ++j)
            acc[i][j] = (v8f){};

    float4 xa[4];

    // ---- x tile: global f32 -> registers (converted to f16 in lstore) ----
    auto gloadA = [&](int kt) {
        const int kk = kt * BK;
#pragma unroll
        for (int i = 0; i < 4; ++i) {                 // 128 rows x 32 f32
            int c   = tid + i * NTHREADS;             // 0..1023 float4-chunks
            int row = c >> 3;
            int c4  = c & 7;
            xa[i] = *(const float4*)(x + (size_t)(m0 + row) * K_TOT + kk + c4 * 4);
        }
    };

    auto lstoreA = [&](int buf) {
#pragma unroll
        for (int i = 0; i < 4; ++i) {
            int c = tid + i * NTHREADS;
            int row = c >> 3, c4 = c & 7;
            union { _Float16 h[4]; uint2 u; } p;
            p.h[0] = (_Float16)xa[i].x; p.h[1] = (_Float16)xa[i].y;
            p.h[2] = (_Float16)xa[i].z; p.h[3] = (_Float16)xa[i].w;
            *(uint2*)&As[buf][row * ASTR + c4 * 4] = p.u;
        }
    };

    // ---- Weff tile: global f16 -> LDS via async copy (no VGPR staging) ----
    // Per thread: 2 x 16-byte chunks of the 128x32-half tile.
    auto asyncB = [&](int kt, int buf) {
        const int kk = kt * BK;
#pragma unroll
        for (int i = 0; i < 2; ++i) {
            int c   = tid + i * NTHREADS;             // 0..511 8-half chunks
            int row = c >> 2;
            int c8  = c & 3;
            unsigned lds_addr =
                (unsigned)(unsigned long long)(uintptr_t)&Bs[buf][row * ASTR + c8 * 8];
            unsigned goff =
                (unsigned)(((n0 + row) * K_TOT + kk + c8 * 8) * (int)sizeof(_Float16));
            asm volatile("global_load_async_to_lds_b128 %0, %1, %2"
                         :
                         : "v"(lds_addr), "v"(goff), "s"(Weff)
                         : "memory");
        }
    };

    // A fragment (16x32 f16): lane holds row m=col; K chunks at hi*8 and 16+hi*8
    auto ldfragA = [&](int buf, int im) -> v16h {
        const _Float16* p = &As[buf][(waveM * 64 + im * 16 + col) * ASTR];
        union { v16h v; uint4 u[2]; } r;
        r.u[0] = *(const uint4*)(p + hi * 8);
        r.u[1] = *(const uint4*)(p + 16 + hi * 8);
        return r.v;
    };
    // B fragment (32x16 f16): lane holds column n=col; K chunks at hi*8 and 16+hi*8
    auto ldfragB = [&](int buf, int jn) -> v16h {
        const _Float16* p = &Bs[buf][(waveN * 32 + jn * 16 + col) * ASTR];
        union { v16h v; uint4 u[2]; } r;
        r.u[0] = *(const uint4*)(p + hi * 8);
        r.u[1] = *(const uint4*)(p + 16 + hi * 8);
        return r.v;
    };

    // Prologue: stage K-step 0.
    asyncB(0, 0);
    gloadA(0);
    lstoreA(0);
    asm volatile("s_wait_asynccnt 0" ::: "memory");
    __syncthreads();

    const int KT = K_TOT / BK;   // 32
#pragma unroll 1
    for (int kt = 0; kt < KT; ++kt) {
        const int buf = kt & 1;
        if (kt + 1 < KT) {
            asyncB(kt + 1, buf ^ 1);   // async copy next B tile straight to LDS
            gloadA(kt + 1);            // overlap x global loads with WMMA
        }

        v16h bf[2];
#pragma unroll
        for (int j = 0; j < 2; ++j) bf[j] = ldfragB(buf, j);
#pragma unroll
        for (int i = 0; i < 4; ++i) {
            v16h af = ldfragA(buf, i);
#pragma unroll
            for (int j = 0; j < 2; ++j)
                acc[i][j] = __builtin_amdgcn_wmma_f32_16x16x32_f16(
                    false, af, false, bf[j], (short)0, acc[i][j], false, false);
        }

        if (kt + 1 < KT) {
            lstoreA(buf ^ 1);
            asm volatile("s_wait_asynccnt 0" ::: "memory");
            __syncthreads();
        }
    }

    // Epilogue: C/D layout -> m = r + 8*hi, n = col; add bias.
#pragma unroll
    for (int j = 0; j < 2; ++j) {
        const int gn = n0 + waveN * 32 + j * 16 + col;
        const float bv = bias[gn];
#pragma unroll
        for (int i = 0; i < 4; ++i) {
            const int gm = m0 + waveM * 64 + i * 16 + hi * 8;
            float* o = out + (size_t)gm * N_TOT + gn;
#pragma unroll
            for (int r = 0; r < 8; ++r)
                o[(size_t)r * N_TOT] = acc[i][j][r] + bv;
        }
    }
}

extern "C" void kernel_launch(void* const* d_in, const int* in_sizes, int n_in,
                              void* d_out, int out_size, void* d_ws, size_t ws_size,
                              hipStream_t stream) {
    const float* x      = (const float*)d_in[0];
    const int*   qW     = (const int*)d_in[1];
    const float* wscale = (const float*)d_in[2];
    const float* bias   = (const float*)d_in[3];
    const float* Amat   = (const float*)d_in[4];
    const float* Bmat   = (const float*)d_in[5];
    float*    out  = (float*)d_out;
    _Float16* Weff = (_Float16*)d_ws;   // 1024*1024*2 = 2 MB scratch

    weff_kernel<<<dim3(K_TOT / 256, N_TOT), 256, 0, stream>>>(qW, wscale, Amat, Bmat, Weff);
    qlora_gemm<<<dim3(N_TOT / BN, M_TOT / BM), NTHREADS, 0, stream>>>(x, Weff, bias, out);
}
